// MyViT_88905823027573
// MI455X (gfx1250) — matrix operand order repaired and checked by reading, not compiled, not run on hardware
//
#include <hip/hip_runtime.h>

// ---------------------------------------------------------------------------
// MyViT forward for gfx1250 (CDNA5, wave32, WMMA bf16 16x16x32).
// DIM=180 padded to 192 (6 K-chunks of 32); HEAD_DIM=15 padded to 16.
// Every WMMA operand (weights, activations, Q/K/V/softmax fragments) is
// produced in fragment-packed order, so all fragment loads are b128 vector
// loads -- zero per-element gather on the consume side.
// ---------------------------------------------------------------------------

typedef __attribute__((ext_vector_type(16))) __bf16          v16bf;
typedef __attribute__((ext_vector_type(16))) unsigned short  us16;
typedef __attribute__((ext_vector_type(8)))  float           v8f;

union FragU { us16 u; v16bf bf; };

__device__ __forceinline__ unsigned short f2bf(float f) {
  union { float f; unsigned u; } v; v.f = f;
  unsigned r = v.u + 0x7FFFu + ((v.u >> 16) & 1u);   // round-to-nearest-even
  return (unsigned short)(r >> 16);
}

// B 32x16 packing used here (contiguous k, not AMD's table order): lanes 0-15 hold K=0..15, lanes 16-31 hold K=16..31
__device__ __forceinline__ int b_k(int lane, int e) {
  return ((lane >> 4) << 4) + e;
}
// Inverse of A 16x32 layout: ksub(0..31) -> (half, e)
__device__ __forceinline__ void inv_a(int ksub, int& half, int& e) {
  half = (ksub >> 4) & 1;
  e = ksub & 15;
}

__device__ __forceinline__ v8f wmma_bf16(v16bf a, v16bf b, v8f c) {
  return __builtin_amdgcn_wmma_f32_16x16x32_bf16(
      false, a, false, b, (short)0, c, false, false);
}

// ---------------------------------------------------------------------------
// Weight prep: fp32 row-major [K][N] -> bf16 fragment-packed
// dst layout: [NT][KC][lane(32)][e(16)]  (32B per lane-fragment)
// ---------------------------------------------------------------------------
__global__ void k_prep_b(const float* __restrict__ src, unsigned short* __restrict__ dst,
                         int K, int N, int KC, int total) {
  int stride = gridDim.x * blockDim.x;
  for (int i = blockIdx.x * blockDim.x + threadIdx.x; i < total; i += stride) {
    int e = i & 15, lane = (i >> 4) & 31;
    int kc = (i >> 9) % KC, nt = i / (KC * 512);
    int n = nt * 16 + (lane & 15);
    int k = kc * 32 + b_k(lane, e);
    float v = (k < K && n < N) ? src[k * N + n] : 0.f;
    dst[i] = f2bf(v);
  }
}

// WQ/WK/WV layer slice [12][180][15] -> fragment-packed [12][6][32][16]
__global__ void k_prep_qkv(const float* __restrict__ src, unsigned short* __restrict__ dst) {
  int stride = gridDim.x * blockDim.x;
  for (int i = blockIdx.x * blockDim.x + threadIdx.x; i < 12 * 6 * 512; i += stride) {
    int e = i & 15, lane = (i >> 4) & 31;
    int kc = (i >> 9) % 6, nt = i / (6 * 512);
    int n = nt * 16 + (lane & 15);
    int k = kc * 32 + b_k(lane, e);
    float v = 0.f;
    if (k < 180 && n < 180) {
      int h = n / 15, he = n % 15;
      v = src[(h * 180 + k) * 15 + he];
    }
    dst[i] = f2bf(v);
  }
}

// ---------------------------------------------------------------------------
// Fused per-frame transformer: patch embed + 6 encoder layers.
// 1 workgroup = 1 frame, 4 waves; wave w owns n-tiles/heads {w, w+4, w+8}.
// Shared memory is one overlaid arena with lifetime-based region reuse:
//   [0      ,12288) Xs   f32[16][192]      residual stream      (whole layer)
//   [12288  ,18432) PaP  us16[6][32]       LN output, A-frags   (LN->GEMM)
//   [18432  ,30720) Qp us16[12][32] -> Ss f32[12][16][16] -> Os f32[16][192]
//   [30720  ,43008) Kp us16[12][32] -> Spack us16[12][32]
//   [43008  ,55296) Vp us16[12][32]
//   [55296  ,55424) mu/rs
// ---------------------------------------------------------------------------
__global__ void __launch_bounds__(128)
k_vit_body(const float* __restrict__ images,
           const float* __restrict__ emb_b,
           const float* __restrict__ pos_emb,
           const float* __restrict__ ln_g,
           const float* __restrict__ ln_b,
           const float* __restrict__ bQ,
           const float* __restrict__ bK,
           const float* __restrict__ bV,
           const float* __restrict__ mlp_b,
           const unsigned short* __restrict__ Wemb,   // [12][6][32][16]
           const unsigned short* __restrict__ WQp,    // x6 layers
           const unsigned short* __restrict__ WKp,
           const unsigned short* __restrict__ WVp,
           const unsigned short* __restrict__ WMp,
           unsigned short* __restrict__ XfP)          // [63][90][32][16] frag-packed
{
  __shared__ __align__(32) char smem[55424];
  float (*Xs)[192]      = (float (*)[192])(smem);
  us16* PaP             = (us16*)(smem + 12288);
  unsigned short* pp    = (unsigned short*)(smem + 12288);
  us16* Qp              = (us16*)(smem + 18432);
  unsigned short* Qps   = (unsigned short*)(smem + 18432);
  float* Ssf            = (float*)(smem + 18432);            // aliases Qp
  float (*Os)[192]      = (float (*)[192])(smem + 18432);    // aliases Ss
  us16* Kp              = (us16*)(smem + 30720);
  unsigned short* Kps   = (unsigned short*)(smem + 30720);
  us16* Sp              = (us16*)(smem + 30720);             // aliases Kp
  unsigned short* Sps   = (unsigned short*)(smem + 30720);
  unsigned short* Vps   = (unsigned short*)(smem + 43008);
  us16* Vp              = (us16*)(smem + 43008);
  float* s_mu           = (float*)(smem + 55296);
  float* s_rs           = (float*)(smem + 55360);

  const int f    = blockIdx.x;
  const int tid  = threadIdx.x;
  const int lane = tid & 31;
  const int wv   = tid >> 5;

  const us16* WembV = (const us16*)Wemb;

  // ---- image -> patches, fragment-packed (all 192 features real)
  for (int i = tid; i < 16 * 192; i += 128) {
    int p = i / 192, j = i % 192;
    int c = j >> 6, hp = (j >> 3) & 7, wp = j & 7;
    int hb = p >> 2, wb = p & 3;
    float v = images[((f * 3 + c) * 32 + hb * 8 + hp) * 32 + wb * 8 + wp];
    int kc = j >> 5, half, e; inv_a(j & 31, half, e);
    pp[((kc * 32) + p + 16 * half) * 16 + e] = f2bf(v);
  }
  __syncthreads();

  // ---- X = P[16,192] @ Wemb + emb_b + pos_emb
  {
    FragU fA[6];
#pragma unroll
    for (int kc = 0; kc < 6; ++kc) fA[kc].u = PaP[kc * 32 + lane];
    for (int nt = wv; nt < 12; nt += 4) {
      v8f acc = {};
#pragma unroll
      for (int kc = 0; kc < 6; ++kc) {
        FragU fb; fb.u = WembV[(nt * 6 + kc) * 32 + lane];
        acc = wmma_bf16(fA[kc].bf, fb.bf, acc);
      }
      int n = nt * 16 + (lane & 15);
#pragma unroll
      for (int r = 0; r < 8; ++r) {
        int m = r + ((lane >> 4) << 3);
        Xs[m][n] = (n < 180) ? acc[r] + emb_b[n] + pos_emb[m * 180 + n] : 0.f;
      }
    }
  }
  __syncthreads();

  const float inv_sqrt_d = 0.074535599f;   // 1/sqrt(180); applied AFTER softmax (ref quirk)

  for (int l = 0; l < 6; ++l) {
    const float* g = ln_g + l * 180;
    const float* b = ln_b + l * 180;
    const us16* wq  = (const us16*)(WQp + l * 36864);
    const us16* wk  = (const us16*)(WKp + l * 36864);
    const us16* wvv = (const us16*)(WVp + l * 36864);
    const us16* wm  = (const us16*)(WMp + l * 36864);

    // ---- LN1 stats
    if (tid < 16) {
      float s = 0.f;
      for (int j = 0; j < 180; ++j) s += Xs[tid][j];
      float mu = s * (1.f / 180.f);
      float var = 0.f;
      for (int j = 0; j < 180; ++j) { float d = Xs[tid][j] - mu; var += d * d; }
      var *= (1.f / 180.f);
      s_mu[tid] = mu; s_rs[tid] = rsqrtf(var + 1e-5f);
    }
    __syncthreads();
    // ---- LN1 -> PaP (frag-packed) ; zero Qp/Kp/Vp regions (36864 B)
    for (int i = tid; i < 16 * 192; i += 128) {
      int p = i / 192, j = i % 192;
      float v = (j < 180) ? (Xs[p][j] - s_mu[p]) * s_rs[p] * g[j] + b[j] : 0.f;
      int kc = j >> 5, half, e; inv_a(j & 31, half, e);
      pp[((kc * 32) + p + 16 * half) * 16 + e] = f2bf(v);
    }
    {
      unsigned long long* z = (unsigned long long*)(smem + 18432);
      for (int i = tid; i < 4608; i += 128) z[i] = 0ull;
    }
    __syncthreads();

    // ---- Q,K,V = nX @ W{q,k,v} + bias, stored directly in attention
    //      fragment layouts: Qp = A-layout (k=he), Kp/Vp = B-layout
    {
      FragU fA[6];
#pragma unroll
      for (int kc = 0; kc < 6; ++kc) fA[kc].u = PaP[kc * 32 + lane];
      for (int nt = wv; nt < 12; nt += 4) {
        v8f aq = {}, ak = {}, av = {};
#pragma unroll
        for (int kc = 0; kc < 6; ++kc) {
          FragU fq, fk, fv;
          fq.u = wq [(nt * 6 + kc) * 32 + lane];
          fk.u = wk [(nt * 6 + kc) * 32 + lane];
          fv.u = wvv[(nt * 6 + kc) * 32 + lane];
          aq = wmma_bf16(fA[kc].bf, fq.bf, aq);
          ak = wmma_bf16(fA[kc].bf, fk.bf, ak);
          av = wmma_bf16(fA[kc].bf, fv.bf, av);
        }
        int n = nt * 16 + (lane & 15);
        if (n < 180) {
          int h = n / 15, he = n % 15;
          float bq = bQ[(l * 12 + h) * 15 + he];
          float bk = bK[(l * 12 + h) * 15 + he];
          float bv_ = bV[(l * 12 + h) * 15 + he];
          int halfA = 0, eA = he;      // he < 15
#pragma unroll
          for (int r = 0; r < 8; ++r) {
            int m = r + ((lane >> 4) << 3);
            // Q: A-layout row m, k=he
            Qps[((h * 32) + m + 16 * halfA) * 16 + eA] = f2bf(aq[r] + bq);
            // K: B-layout col n=m(patch), k=he (k<16 -> lanes 0-15, e=he)
            Kps[((h * 32) + m) * 16 + he] = f2bf(ak[r] + bk);
            // V: B-layout col n=he, k=m(patch)  (k<16 -> lanes 0-15, e=m)
            Vps[((h * 32) + he) * 16 + m] = f2bf(av[r] + bv_);
          }
        }
      }
    }
    __syncthreads();

    // ---- scores S_h = Q_h @ K_h^T : one WMMA per head, vector-loaded frags.
    //      Accumulators staged in registers so Ss can overwrite Qp.
    v8f sacc[3];
    {
      int hi = 0;
      for (int h = wv; h < 12; h += 4, ++hi) {
        FragU fa, fb;
        fa.u = Qp[h * 32 + lane];
        fb.u = Kp[h * 32 + lane];
        v8f z = {};
        sacc[hi] = wmma_bf16(fa.bf, fb.bf, z);
      }
    }
    __syncthreads();                       // Qp/Kp dead everywhere
    {
      int hi = 0;
      for (int h = wv; h < 12; h += 4, ++hi) {
        int n = lane & 15;
#pragma unroll
        for (int r = 0; r < 8; ++r)
          Ssf[(h * 16 + r + ((lane >> 4) << 3)) * 16 + n] = sacc[hi][r];
      }
      // zero Spack region (aliases Kp) so padded K-slots (k>=16) are zero
      unsigned long long* z = (unsigned long long*)Sp;
      for (int i = tid; i < 1536; i += 128) z[i] = 0ull;
    }
    __syncthreads();

    // ---- softmax over k(16), scale by 1/sqrt(180), write A-layout Spack
    for (int rr = tid; rr < 192; rr += 128) {
      int h = rr >> 4, m = rr & 15;
      float* row = Ssf + (h * 16 + m) * 16;
      float mx = -1e30f;
#pragma unroll
      for (int j = 0; j < 16; ++j) mx = fmaxf(mx, row[j]);
      float vv[16], s = 0.f;
#pragma unroll
      for (int j = 0; j < 16; ++j) { vv[j] = __expf(row[j] - mx); s += vv[j]; }
      float inv = inv_sqrt_d / s;
#pragma unroll
      for (int j = 0; j < 16; ++j)
        Sps[((h * 32) + m) * 16 + j] = f2bf(vv[j] * inv);
    }
    __syncthreads();

    // ---- O_h = softmax(S)_h @ V_h : one WMMA per head; Os overwrites Ss
    for (int h = wv; h < 12; h += 4) {
      FragU fa, fb;
      fa.u = Sp[h * 32 + lane];
      fb.u = Vp[h * 32 + lane];
      v8f z = {};
      v8f acc = wmma_bf16(fa.bf, fb.bf, z);
      int n = lane & 15;
      if (n < 15) {
#pragma unroll
        for (int r = 0; r < 8; ++r) Os[r + ((lane >> 4) << 3)][h * 15 + n] = acc[r];
      }
    }
    __syncthreads();

    // ---- residual + LN2 -> Xs (f32) and PaP (frag-packed)
    if (tid < 16) {
      float s = 0.f;
      for (int j = 0; j < 180; ++j) s += Os[tid][j] + Xs[tid][j];
      float mu = s * (1.f / 180.f);
      float var = 0.f;
      for (int j = 0; j < 180; ++j) { float d = Os[tid][j] + Xs[tid][j] - mu; var += d * d; }
      var *= (1.f / 180.f);
      s_mu[tid] = mu; s_rs[tid] = rsqrtf(var + 1e-5f);
    }
    __syncthreads();
    for (int i = tid; i < 16 * 192; i += 128) {
      int p = i / 192, j = i % 192;
      float v = (j < 180) ? (Os[p][j] + Xs[p][j] - s_mu[p]) * s_rs[p] * g[j] + b[j] : 0.f;
      Xs[p][j] = v;
      int kc = j >> 5, half, e; inv_a(j & 31, half, e);
      pp[((kc * 32) + p + 16 * half) * 16 + e] = f2bf(v);
    }
    __syncthreads();

    // ---- MLP: Xs = relu(nX2 @ mlp_W + mlp_b)   (no residual, per source)
    {
      FragU fA[6];
#pragma unroll
      for (int kc = 0; kc < 6; ++kc) fA[kc].u = PaP[kc * 32 + lane];
      for (int nt = wv; nt < 12; nt += 4) {
        v8f acc = {};
#pragma unroll
        for (int kc = 0; kc < 6; ++kc) {
          FragU fb; fb.u = wm[(nt * 6 + kc) * 32 + lane];
          acc = wmma_bf16(fA[kc].bf, fb.bf, acc);
        }
        int n = nt * 16 + (lane & 15);
        float bb = (n < 180) ? mlp_b[l * 180 + n] : 0.f;
#pragma unroll
        for (int r = 0; r < 8; ++r) {
          int m = r + ((lane >> 4) << 3);
          Xs[m][n] = (n < 180) ? fmaxf(acc[r] + bb, 0.f) : 0.f;
        }
      }
    }
    __syncthreads();
  }

  // ---- write features bf16 fragment-packed for head-l1 A operand
  {
    int ft = f >> 4, m = f & 15;
    for (int i = tid; i < 16 * 180; i += 128) {
      int p = i / 180, j = i % 180;
      int q = p * 180 + j;
      int kc = q >> 5, half, e; inv_a(q & 31, half, e);
      XfP[(((size_t)ft * 90 + kc) * 32 + m + 16 * half) * 16 + e] = f2bf(Xs[p][j]);
    }
  }
}

// ---------------------------------------------------------------------------
// Head: l1 [1000,2880]@[2880,180] relu -> l2 @[180,180] relu -> l3 @[180,10]
// + softmax. M-tile = 16 frames per wave. All operands fragment-packed bf16.
// ---------------------------------------------------------------------------
__global__ void __launch_bounds__(32)
k_head1(const unsigned short* __restrict__ XfP,   // [63][90][32][16]
        const unsigned short* __restrict__ W1p,   // [12][90][32][16]
        const float* __restrict__ b1,
        unsigned short* __restrict__ H1P)         // [63][6][32][16]
{
  const int nt = blockIdx.x;            // 0..11
  const int ft = blockIdx.y;            // 0..62
  const int lane = threadIdx.x;
  const us16* Av = (const us16*)XfP;
  const us16* Bv = (const us16*)W1p;
  v8f acc = {};
  for (int kc = 0; kc < 90; ++kc) {
    FragU fa, fb;
    fa.u = Av[(ft * 90 + kc) * 32 + lane];
    fb.u = Bv[(nt * 90 + kc) * 32 + lane];
    if (kc + 4 < 90) {   // stream-ahead prefetch of the weight stream
      __builtin_prefetch((const void*)&Bv[(nt * 90 + kc + 4) * 32 + lane], 0, 0);
    }
    acc = wmma_bf16(fa.bf, fb.bf, acc);
  }
  int n = nt * 16 + (lane & 15);
  float bb = (n < 180) ? b1[n] : 0.f;
  int kc2 = n >> 5, half2, e2; inv_a(n & 31, half2, e2);
#pragma unroll
  for (int r = 0; r < 8; ++r) {
    int m = r + ((lane >> 4) << 3);
    float v = (n < 180) ? fmaxf(acc[r] + bb, 0.f) : 0.f;
    H1P[(((size_t)ft * 6 + kc2) * 32 + m + 16 * half2) * 16 + e2] = f2bf(v);
  }
}

__global__ void __launch_bounds__(32)
k_head2(const unsigned short* __restrict__ H1P,   // [63][6][32][16]
        const unsigned short* __restrict__ W2p,   // [12][6][32][16]
        const float* __restrict__ b2,
        unsigned short* __restrict__ H2P)         // [63][6][32][16]
{
  const int nt = blockIdx.x;
  const int ft = blockIdx.y;
  const int lane = threadIdx.x;
  const us16* Av = (const us16*)H1P;
  const us16* Bv = (const us16*)W2p;
  v8f acc = {};
#pragma unroll
  for (int kc = 0; kc < 6; ++kc) {
    FragU fa, fb;
    fa.u = Av[(ft * 6 + kc) * 32 + lane];
    fb.u = Bv[(nt * 6 + kc) * 32 + lane];
    acc = wmma_bf16(fa.bf, fb.bf, acc);
  }
  int n = nt * 16 + (lane & 15);
  float bb = (n < 180) ? b2[n] : 0.f;
  int kc2 = n >> 5, half2, e2; inv_a(n & 31, half2, e2);
#pragma unroll
  for (int r = 0; r < 8; ++r) {
    int m = r + ((lane >> 4) << 3);
    float v = (n < 180) ? fmaxf(acc[r] + bb, 0.f) : 0.f;
    H2P[(((size_t)ft * 6 + kc2) * 32 + m + 16 * half2) * 16 + e2] = f2bf(v);
  }
}

__global__ void __launch_bounds__(32)
k_head3(const unsigned short* __restrict__ H2P,   // [63][6][32][16]
        const unsigned short* __restrict__ W3p,   // [1][6][32][16]
        const float* __restrict__ b3,
        float* __restrict__ out)                  // [1000][10]
{
  __shared__ float L[16][16];
  const int ft = blockIdx.x;
  const int lane = threadIdx.x;
  const us16* Av = (const us16*)H2P;
  const us16* Bv = (const us16*)W3p;
  v8f acc = {};
#pragma unroll
  for (int kc = 0; kc < 6; ++kc) {
    FragU fa, fb;
    fa.u = Av[(ft * 6 + kc) * 32 + lane];
    fb.u = Bv[kc * 32 + lane];
    acc = wmma_bf16(fa.bf, fb.bf, acc);
  }
  int n = lane & 15;
#pragma unroll
  for (int r = 0; r < 8; ++r) {
    L[r + ((lane >> 4) << 3)][n] = acc[r] + ((n < 10) ? b3[n] : 0.f);
  }
  __syncthreads();
  if (lane < 16) {
    int fr = ft * 16 + lane;
    if (fr < 1000) {
      float mx = -1e30f;
      for (int j = 0; j < 10; ++j) mx = fmaxf(mx, L[lane][j]);
      float s = 0.f;
      for (int j = 0; j < 10; ++j) { float e = __expf(L[lane][j] - mx); L[lane][j] = e; s += e; }
      float inv = 1.f / s;
      for (int j = 0; j < 10; ++j) out[fr * 10 + j] = L[lane][j] * inv;
    }
  }
}

// ---------------------------------------------------------------------------
extern "C" void kernel_launch(void* const* d_in, const int* in_sizes, int n_in,
                              void* d_out, int out_size, void* d_ws, size_t ws_size,
                              hipStream_t stream)
{
  const float* images = (const float*)d_in[0];
  const float* emb_W  = (const float*)d_in[1];
  const float* emb_b  = (const float*)d_in[2];
  const float* pos    = (const float*)d_in[3];
  const float* ln_g   = (const float*)d_in[4];
  const float* ln_bb  = (const float*)d_in[5];
  const float* WQ     = (const float*)d_in[6];
  const float* bQ     = (const float*)d_in[7];
  const float* WK     = (const float*)d_in[8];
  const float* bK     = (const float*)d_in[9];
  const float* WV     = (const float*)d_in[10];
  const float* bV     = (const float*)d_in[11];
  const float* mlp_W  = (const float*)d_in[12];
  const float* mlp_b  = (const float*)d_in[13];
  const float* l1_W   = (const float*)d_in[14];
  const float* l1_b   = (const float*)d_in[15];
  const float* l2_W   = (const float*)d_in[16];
  const float* l2_b   = (const float*)d_in[17];
  const float* l3_W   = (const float*)d_in[18];
  const float* l3_b   = (const float*)d_in[19];
  float* out = (float*)d_out;

  char* ws = (char*)d_ws;
  size_t off = 0;
  auto alloc = [&](size_t bytes) -> char* {
    char* p = ws + off;
    off = (off + bytes + 255) & ~(size_t)255;
    return p;
  };
  unsigned short* XfP  = (unsigned short*)alloc((size_t)63 * 90 * 512 * 2);
  unsigned short* H1P  = (unsigned short*)alloc((size_t)63 * 6 * 512 * 2);
  unsigned short* H2P  = (unsigned short*)alloc((size_t)63 * 6 * 512 * 2);
  unsigned short* Wemb = (unsigned short*)alloc((size_t)12 * 6 * 512 * 2);
  unsigned short* WQp  = (unsigned short*)alloc((size_t)6 * 12 * 6 * 512 * 2);
  unsigned short* WKp  = (unsigned short*)alloc((size_t)6 * 12 * 6 * 512 * 2);
  unsigned short* WVp  = (unsigned short*)alloc((size_t)6 * 12 * 6 * 512 * 2);
  unsigned short* WMp  = (unsigned short*)alloc((size_t)6 * 12 * 6 * 512 * 2);
  unsigned short* WL1  = (unsigned short*)alloc((size_t)12 * 90 * 512 * 2);
  unsigned short* WL2  = (unsigned short*)alloc((size_t)12 * 6 * 512 * 2);
  unsigned short* WL3  = (unsigned short*)alloc((size_t)1 * 6 * 512 * 2);

  // ---- weight prep (bf16, fragment-packed) -- tiny, L2-resident afterwards
  k_prep_b<<<144, 256, 0, stream>>>(emb_W, Wemb, 192, 180, 6, 12 * 6 * 512);
  for (int l = 0; l < 6; ++l) {
    k_prep_qkv<<<144, 256, 0, stream>>>(WQ + (size_t)l * 32400, WQp + (size_t)l * 36864);
    k_prep_qkv<<<144, 256, 0, stream>>>(WK + (size_t)l * 32400, WKp + (size_t)l * 36864);
    k_prep_qkv<<<144, 256, 0, stream>>>(WV + (size_t)l * 32400, WVp + (size_t)l * 36864);
    k_prep_b<<<144, 256, 0, stream>>>(mlp_W + (size_t)l * 32400, WMp + (size_t)l * 36864,
                                      180, 180, 6, 12 * 6 * 512);
  }
  k_prep_b<<<2160, 256, 0, stream>>>(l1_W, WL1, 2880, 180, 90, 12 * 90 * 512);
  k_prep_b<<<144, 256, 0, stream>>>(l2_W, WL2, 180, 180, 6, 12 * 6 * 512);
  k_prep_b<<<12, 256, 0, stream>>>(l3_W, WL3, 180, 10, 6, 1 * 6 * 512);

  // ---- fused transformer: one workgroup per frame
  k_vit_body<<<1000, 128, 0, stream>>>(images, emb_b, pos, ln_g, ln_bb,
                                       bQ, bK, bV, mlp_b,
                                       Wemb, WQp, WKp, WVp, WMp, XfP);

  // ---- MLP head
  dim3 gh(12, 63);
  k_head1<<<gh, 32, 0, stream>>>(XfP, WL1, l1_b, H1P);
  k_head2<<<gh, 32, 0, stream>>>(H1P, WL2, l2_b, H2P);
  k_head3<<<63, 32, 0, stream>>>(H2P, WL3, l3_b, out);

  (void)in_sizes; (void)n_in; (void)out_size; (void)ws_size;
}
